// MultiHeadAttention_58548994179754
// MI455X (gfx1250) — matrix operand order repaired and checked by reading, not compiled, not run on hardware
//
#include <hip/hip_runtime.h>

#define B_SZ   2
#define N_SEQ  2048
#define E_DIM  1024
#define H_NUM  16
#define D_HEAD 64
#define SCALE_F 0.03125f   // EMBED_DIM ** -0.5

typedef __attribute__((ext_vector_type(16))) _Float16 v16h;
typedef __attribute__((ext_vector_type(8)))  _Float16 v8h;
typedef __attribute__((ext_vector_type(4)))  _Float16 v4h;
typedef __attribute__((ext_vector_type(8)))  float    v8f;
typedef __attribute__((ext_vector_type(4)))  unsigned u32x4;
typedef __attribute__((ext_vector_type(4)))  int      i32x4;
typedef __attribute__((ext_vector_type(8)))  int      i32x8;

// ---------------------------------------------------------------------------
// WMMA helper: D = A(16x32 f16) * B(32x16 f16) + C(16x16 f32)
// ---------------------------------------------------------------------------
static __device__ __forceinline__ v8f wmma16(v16h a, v16h b, v8f c) {
  return __builtin_amdgcn_wmma_f32_16x16x32_f16(false, a, false, b,
                                                (short)0, c, false, false);
}

static __device__ __forceinline__ v16h mk16(v8h a, v8h b) {
  v16h r;
#pragma unroll
  for (int i = 0; i < 8; ++i) { r[i] = a[i]; r[i + 8] = b[i]; }
  return r;
}

// ---------------------------------------------------------------------------
// TDM: async 2D tile load (f16 elements) global -> LDS. Descriptor built per
// CDNA5 ISA (D# group0/group1). Tracked by TENSORcnt.
// ---------------------------------------------------------------------------
static __device__ __forceinline__ void tdm_load_2d_f16(
    unsigned lds_off, const _Float16* gptr,
    unsigned tile_d0, unsigned tile_d1, unsigned stride0_elems)
{
  unsigned long long ga = (unsigned long long)(size_t)gptr;
  u32x4 g0;
  g0[0] = 1u;                                          // count=1, user D#
  g0[1] = lds_off;                                     // lds_addr (bytes)
  g0[2] = (unsigned)(ga & 0xFFFFFFFFull);              // global_addr lo
  g0[3] = (unsigned)((ga >> 32) & 0x1FFFFFFull) | (2u << 30); // hi + type=2

  const unsigned td0 = 0x100000u, td1 = 0x100000u;     // huge dims: no OOB clip
  i32x8 g1;
  g1[0] = (int)(1u << 16);                             // data_size = 2 bytes
  g1[1] = (int)((td0 & 0xFFFFu) << 16);                // tensor_dim0 lo16
  g1[2] = (int)(((td0 >> 16) & 0xFFFFu) | ((td1 & 0xFFFFu) << 16));
  g1[3] = (int)(((td1 >> 16) & 0xFFFFu) | ((tile_d0 & 0xFFFFu) << 16));
  g1[4] = (int)(tile_d1 & 0xFFFFu);                    // tile_dim1 (tile_dim2=0)
  g1[5] = (int)stride0_elems;                          // tensor_dim0_stride lo32
  g1[6] = 0;                                           // stride hi + dim1_stride
  g1[7] = 0;
  i32x4 z4 = {0, 0, 0, 0};
#if defined(__clang_major__) && (__clang_major__ >= 23)
  i32x8 z8 = {0, 0, 0, 0, 0, 0, 0, 0};
  __builtin_amdgcn_tensor_load_to_lds(g0, g1, z4, z4, z8, 0);
#else
  __builtin_amdgcn_tensor_load_to_lds(g0, g1, z4, z4, 0);
#endif
}

// ---------------------------------------------------------------------------
// fp32 -> f16 conversion (vectorized x4)
// ---------------------------------------------------------------------------
__global__ __launch_bounds__(256) void cvt_f32_f16(const float4* __restrict__ s,
                                                   v4h* __restrict__ d, int n4) {
  int i = blockIdx.x * blockDim.x + threadIdx.x;
  if (i < n4) {
    float4 f = s[i];
    v4h h; h[0] = (_Float16)f.x; h[1] = (_Float16)f.y;
           h[2] = (_Float16)f.z; h[3] = (_Float16)f.w;
    d[i] = h;
  }
}

// ---------------------------------------------------------------------------
// Generic WMMA GEMM:  C[M x Ncols] = A[M x K] * W[Ncols x K]^T + bias
// WG = 256 threads (8 waves); WG tile = 128 rows x 64 cols; wave tile 16x64.
// Inner loop: ping-pong register double-buffering (unroll x2, no reg copies).
// ---------------------------------------------------------------------------
__global__ __launch_bounds__(256) void gemm_wmma(
    const _Float16* __restrict__ A, const _Float16* __restrict__ W,
    const float* __restrict__ bias,
    _Float16* __restrict__ out1, _Float16* __restrict__ out2,
    float* __restrict__ outf,
    int K, int Ncols, int split, float scale)
{
  const int w    = threadIdx.x >> 5;
  const int lane = threadIdx.x & 31;
  const int lo   = lane & 15;
  const int hi   = lane >> 4;
  const int wgRow = blockIdx.y * 128;
  const int wgCol = blockIdx.x * 64;

  const int r = wgRow + w * 16 + lo;               // A row for this lane (m=lo)
  const _Float16* arow = A + (size_t)r * K + 16 * hi;
  const _Float16* wr0 = W + (size_t)(wgCol +  0 + lo) * K + 16 * hi;
  const _Float16* wr1 = W + (size_t)(wgCol + 16 + lo) * K + 16 * hi;
  const _Float16* wr2 = W + (size_t)(wgCol + 32 + lo) * K + 16 * hi;
  const _Float16* wr3 = W + (size_t)(wgCol + 48 + lo) * K + 16 * hi;

  v8f acc[4] = {v8f{}, v8f{}, v8f{}, v8f{}};

  v16h Af0, Bf0[4], Af1, Bf1[4];

  // preload k-step 0 into set 0
  Af0    = mk16(*(const v8h*)(arow),    *(const v8h*)(arow + 8));
  Bf0[0] = mk16(*(const v8h*)(wr0),     *(const v8h*)(wr0 + 8));
  Bf0[1] = mk16(*(const v8h*)(wr1),     *(const v8h*)(wr1 + 8));
  Bf0[2] = mk16(*(const v8h*)(wr2),     *(const v8h*)(wr2 + 8));
  Bf0[3] = mk16(*(const v8h*)(wr3),     *(const v8h*)(wr3 + 8));

  // ping-pong main loop: processes k-steps kb (set0) and kb+32 (set1)
  int kb = 0;
  for (; kb + 64 < K; kb += 64) {
    const int k1 = kb + 32, k2 = kb + 64;
    // load set1 (k1) while computing with set0 (kb)
    Af1    = mk16(*(const v8h*)(arow + k1), *(const v8h*)(arow + k1 + 8));
    Bf1[0] = mk16(*(const v8h*)(wr0 + k1),  *(const v8h*)(wr0 + k1 + 8));
    Bf1[1] = mk16(*(const v8h*)(wr1 + k1),  *(const v8h*)(wr1 + k1 + 8));
    Bf1[2] = mk16(*(const v8h*)(wr2 + k1),  *(const v8h*)(wr2 + k1 + 8));
    Bf1[3] = mk16(*(const v8h*)(wr3 + k1),  *(const v8h*)(wr3 + k1 + 8));
#pragma unroll
    for (int t = 0; t < 4; ++t) acc[t] = wmma16(Af0, Bf0[t], acc[t]);
    // load set0 (k2) while computing with set1 (k1)
    Af0    = mk16(*(const v8h*)(arow + k2), *(const v8h*)(arow + k2 + 8));
    Bf0[0] = mk16(*(const v8h*)(wr0 + k2),  *(const v8h*)(wr0 + k2 + 8));
    Bf0[1] = mk16(*(const v8h*)(wr1 + k2),  *(const v8h*)(wr1 + k2 + 8));
    Bf0[2] = mk16(*(const v8h*)(wr2 + k2),  *(const v8h*)(wr2 + k2 + 8));
    Bf0[3] = mk16(*(const v8h*)(wr3 + k2),  *(const v8h*)(wr3 + k2 + 8));
#pragma unroll
    for (int t = 0; t < 4; ++t) acc[t] = wmma16(Af1, Bf1[t], acc[t]);
  }
  // drain: two k-steps remain (kb and kb+32)
  {
    const int k1 = kb + 32;
    Af1    = mk16(*(const v8h*)(arow + k1), *(const v8h*)(arow + k1 + 8));
    Bf1[0] = mk16(*(const v8h*)(wr0 + k1),  *(const v8h*)(wr0 + k1 + 8));
    Bf1[1] = mk16(*(const v8h*)(wr1 + k1),  *(const v8h*)(wr1 + k1 + 8));
    Bf1[2] = mk16(*(const v8h*)(wr2 + k1),  *(const v8h*)(wr2 + k1 + 8));
    Bf1[3] = mk16(*(const v8h*)(wr3 + k1),  *(const v8h*)(wr3 + k1 + 8));
#pragma unroll
    for (int t = 0; t < 4; ++t) acc[t] = wmma16(Af0, Bf0[t], acc[t]);
#pragma unroll
    for (int t = 0; t < 4; ++t) acc[t] = wmma16(Af1, Bf1[t], acc[t]);
  }

  // --- epilogue -------------------------------------------------------------
#pragma unroll
  for (int t = 0; t < 4; ++t) {
    const int c = wgCol + t * 16 + lo;
    const float bv = bias[c];
#pragma unroll
    for (int v = 0; v < 8; ++v) {
      const int row = wgRow + w * 16 + v + 8 * hi;
      const float val = (acc[t][v] + bv) * scale;
      if (outf) {
        outf[(size_t)row * split + c] = val;
      } else if (c < split) {
        out1[(size_t)row * split + c] = (_Float16)val;
      } else {
        // V transposed: xvT[(b*E + (c-split)) * N + n]
        const int bb = row / N_SEQ, nn = row % N_SEQ;
        out2[((size_t)bb * split + (c - split)) * N_SEQ + nn] = (_Float16)val;
      }
    }
  }
}

// ---------------------------------------------------------------------------
// Flash attention with add-one softmax (== softmax over [logits, 0]).
// Grid: B*H * (N/128) WGs; 256 threads = 8 waves; each wave owns 16 q-rows.
// Keys streamed in 64-row blocks via TDM double-buffering in LDS.
// ---------------------------------------------------------------------------
__global__ __launch_bounds__(256) void attn_wmma(
    const _Float16* __restrict__ xq,   // (B,N,E) pre-scaled by SCALE
    const _Float16* __restrict__ xk,   // (B,N,E)
    const _Float16* __restrict__ xvT,  // (B,E,N)  d-major
    const int* __restrict__ mask,      // (B,N)
    _Float16* __restrict__ attn)       // (B,N,E)
{
  __shared__ _Float16 Kbuf[2][64 * 64];   // [key][d]
  __shared__ _Float16 Vbuf[2][64 * 64];   // [d][key]
  __shared__ _Float16 Pst[8][16 * 32];    // per-wave P staging

  const int bh   = blockIdx.x % (B_SZ * H_NUM);
  const int qblk = blockIdx.x / (B_SZ * H_NUM);
  const int b = bh / H_NUM, h = bh % H_NUM;
  const int w    = threadIdx.x >> 5;
  const int lane = threadIdx.x & 31;
  const int lo   = lane & 15;
  const int hi   = lane >> 4;

  // --- Q fragments (2 k-steps over D=64) -----------------------------------
  const _Float16* qrow =
      xq + ((size_t)(b * N_SEQ + qblk * 128 + w * 16 + lo)) * E_DIM + h * D_HEAD;
  v16h Qa[2];
#pragma unroll
  for (int kk = 0; kk < 2; ++kk) {
    v8h x0 = *(const v8h*)(qrow + kk * 32 + 16 * hi);
    v8h x1 = *(const v8h*)(qrow + kk * 32 + 16 * hi + 8);
    Qa[kk] = mk16(x0, x1);
  }

  v8f O[4] = {v8f{}, v8f{}, v8f{}, v8f{}};
  float rmax[8], rsum[8];
#pragma unroll
  for (int v = 0; v < 8; ++v) { rmax[v] = 0.0f; rsum[v] = 1.0f; }  // phantom "+1"

  const int NBLK = N_SEQ / 64;   // 32 key blocks

  // prime first TDM loads (wave 0 issues; TENSORcnt tracks)
  if (w == 0) {
    tdm_load_2d_f16((unsigned)(size_t)&Kbuf[0][0],
                    xk + ((size_t)(b * N_SEQ)) * E_DIM + h * D_HEAD,
                    64, 64, E_DIM);
    tdm_load_2d_f16((unsigned)(size_t)&Vbuf[0][0],
                    xvT + ((size_t)b * E_DIM + h * D_HEAD) * N_SEQ,
                    64, 64, N_SEQ);
  }

  for (int kb = 0; kb < NBLK; ++kb) {
    const int buf = kb & 1;
    if (w == 0) __builtin_amdgcn_s_wait_tensorcnt(0);
    __syncthreads();                         // block kb resident in LDS
    if (w == 0 && (kb + 1) < NBLK) {         // prefetch kb+1 into other buffer
      const int nk = kb + 1;
      tdm_load_2d_f16((unsigned)(size_t)&Kbuf[buf ^ 1][0],
                      xk + ((size_t)(b * N_SEQ + nk * 64)) * E_DIM + h * D_HEAD,
                      64, 64, E_DIM);
      tdm_load_2d_f16((unsigned)(size_t)&Vbuf[buf ^ 1][0],
                      xvT + ((size_t)b * E_DIM + h * D_HEAD) * N_SEQ + nk * 64,
                      64, 64, N_SEQ);
    }
    const _Float16* Kl = &Kbuf[buf][0];
    const _Float16* Vl = &Vbuf[buf][0];

#pragma unroll
    for (int j32 = 0; j32 < 2; ++j32) {      // two 32-key sub-blocks
      // ---- S = Q @ K^T (16x32 tile as two 16x16 f32 tiles) ----------------
      v8f S0 = {}, S1 = {};
#pragma unroll
      for (int kk = 0; kk < 2; ++kk) {
        const _Float16* k0 = Kl + (j32 * 32 + lo)      * 64 + kk * 32 + 16 * hi;
        const _Float16* k1 = Kl + (j32 * 32 + 16 + lo) * 64 + kk * 32 + 16 * hi;
        v16h B0 = mk16(*(const v8h*)k0, *(const v8h*)(k0 + 8));
        v16h B1 = mk16(*(const v8h*)k1, *(const v8h*)(k1 + 8));
        S0 = wmma16(Qa[kk], B0, S0);
        S1 = wmma16(Qa[kk], B1, S1);
      }
      // ---- mask add: +float(mask) or -inf ---------------------------------
      const int key0 = kb * 64 + j32 * 32;
      const int m0 = mask[b * N_SEQ + key0 + lo];
      const int m1 = mask[b * N_SEQ + key0 + 16 + lo];
      const float a0 = (m0 == 0) ? -__builtin_inff() : (float)m0;
      const float a1 = (m1 == 0) ? -__builtin_inff() : (float)m1;
#pragma unroll
      for (int v = 0; v < 8; ++v) { S0[v] += a0; S1[v] += a1; }

      // ---- online softmax update (phantom element handles "+exp(-m)") -----
      float corr[8];
#pragma unroll
      for (int v = 0; v < 8; ++v) {
        float bm = fmaxf(S0[v], S1[v]);
        bm = fmaxf(bm, __shfl_xor(bm, 1));
        bm = fmaxf(bm, __shfl_xor(bm, 2));
        bm = fmaxf(bm, __shfl_xor(bm, 4));
        bm = fmaxf(bm, __shfl_xor(bm, 8));
        const float nm = fmaxf(rmax[v], bm);
        corr[v] = __expf(rmax[v] - nm);
        rmax[v] = nm;
        const float p0 = __expf(S0[v] - nm);
        const float p1 = __expf(S1[v] - nm);
        S0[v] = p0; S1[v] = p1;
        float bs = p0 + p1;
        bs += __shfl_xor(bs, 1);
        bs += __shfl_xor(bs, 2);
        bs += __shfl_xor(bs, 4);
        bs += __shfl_xor(bs, 8);
        rsum[v] = rsum[v] * corr[v] + bs;
      }
#pragma unroll
      for (int t = 0; t < 4; ++t)
#pragma unroll
        for (int v = 0; v < 8; ++v) O[t][v] *= corr[v];

      // ---- P: C-layout -> A-layout via per-wave LDS staging ---------------
      _Float16* ps = &Pst[w][0];
#pragma unroll
      for (int v = 0; v < 8; ++v) {
        ps[(v + 8 * hi) * 32 + lo]      = (_Float16)S0[v];
        ps[(v + 8 * hi) * 32 + 16 + lo] = (_Float16)S1[v];
      }
      asm volatile("" ::: "memory");   // keep LDS store->load order (per-wave)
      v16h Pa = mk16(*(const v8h*)(ps + lo * 32 + 16 * hi),
                     *(const v8h*)(ps + lo * 32 + 16 * hi + 8));

      // ---- O += P @ V (V^T in LDS: contiguous B fragments) ----------------
#pragma unroll
      for (int t = 0; t < 4; ++t) {
        const _Float16* vr = Vl + (t * 16 + lo) * 64 + j32 * 32 + 16 * hi;
        v16h Bv = mk16(*(const v8h*)vr, *(const v8h*)(vr + 8));
        O[t] = wmma16(Pa, Bv, O[t]);
      }
    }
    __syncthreads();                        // all waves done with buf
  }

  // --- finalize: divide by denominator, store f16 (B,N,E) -------------------
  const size_t obase =
      ((size_t)(b * N_SEQ + qblk * 128 + w * 16)) * E_DIM + h * D_HEAD;
#pragma unroll
  for (int v = 0; v < 8; ++v) {
    const float inv = 1.0f / rsum[v];
#pragma unroll
    for (int t = 0; t < 4; ++t)
      attn[obase + (size_t)(v + 8 * hi) * E_DIM + t * 16 + lo] =
          (_Float16)(O[t][v] * inv);
  }
}

// ---------------------------------------------------------------------------
// Host launcher
// ---------------------------------------------------------------------------
extern "C" void kernel_launch(void* const* d_in, const int* in_sizes, int n_in,
                              void* d_out, int out_size, void* d_ws, size_t ws_size,
                              hipStream_t stream) {
  (void)in_sizes; (void)n_in; (void)out_size; (void)ws_size;
  const float* q    = (const float*)d_in[0];
  const float* kv   = (const float*)d_in[1];
  const int*   mask = (const int*)d_in[2];
  const float* Wq   = (const float*)d_in[3];
  const float* bq   = (const float*)d_in[4];
  const float* Wkv  = (const float*)d_in[5];
  const float* bkv  = (const float*)d_in[6];
  const float* Wo   = (const float*)d_in[7];
  const float* bo   = (const float*)d_in[8];
  float* out = (float*)d_out;

  const size_t BNE = (size_t)B_SZ * N_SEQ * E_DIM;   // 4,194,304
  const size_t EE  = (size_t)E_DIM * E_DIM;          // 1,048,576
  _Float16* wsh  = (_Float16*)d_ws;
  _Float16* qh   = wsh;
  _Float16* kvh  = qh   + BNE;
  _Float16* Wqh  = kvh  + BNE;
  _Float16* Wkvh = Wqh  + EE;
  _Float16* Woh  = Wkvh + 2 * EE;
  _Float16* xqh  = Woh  + EE;
  _Float16* xkh  = xqh  + BNE;
  _Float16* xvTh = xkh  + BNE;
  _Float16* ath  = xvTh + BNE;

  auto cvt = [&](const float* s, _Float16* d, size_t n) {
    int n4 = (int)(n / 4);
    cvt_f32_f16<<<dim3((n4 + 255) / 256), dim3(256), 0, stream>>>(
        (const float4*)s, (v4h*)d, n4);
  };
  cvt(q,   qh,   BNE);
  cvt(kv,  kvh,  BNE);
  cvt(Wq,  Wqh,  EE);
  cvt(Wkv, Wkvh, 2 * EE);
  cvt(Wo,  Woh,  EE);

  const int M = B_SZ * N_SEQ;  // 4096
  // xq = (q @ Wq^T + bq) * SCALE  -> f16 (B,N,E)
  gemm_wmma<<<dim3(E_DIM / 64, M / 128), dim3(256), 0, stream>>>(
      qh, Wqh, bq, xqh, nullptr, nullptr, E_DIM, E_DIM, E_DIM, SCALE_F);
  // xkv = kv @ Wkv^T + bkv -> xk f16 (B,N,E), xv f16 transposed (B,E,N)
  gemm_wmma<<<dim3(2 * E_DIM / 64, M / 128), dim3(256), 0, stream>>>(
      kvh, Wkvh, bkv, xkh, xvTh, nullptr, E_DIM, 2 * E_DIM, E_DIM, 1.0f);
  // attention
  attn_wmma<<<dim3(B_SZ * H_NUM * (N_SEQ / 128)), dim3(256), 0, stream>>>(
      xqh, xkh, xvTh, mask, ath);
  // out = attn @ Wo^T + bo -> f32
  gemm_wmma<<<dim3(E_DIM / 64, M / 128), dim3(256), 0, stream>>>(
      ath, Woh, bo, nullptr, nullptr, out, E_DIM, E_DIM, E_DIM, 1.0f);
}
